// GCNModel_8546984919421
// MI455X (gfx1250) — matrix operand order repaired and checked by reading, not compiled, not run on hardware
//
#include <hip/hip_runtime.h>

// ---------------------------------------------------------------------------
// 3-layer GCN on gfx1250 (wave32).
//   L1: h1 = relu(scatter(ew * (x@W1)[src] -> dst) + b1)      K=2   (scalar FMA)
//   L2: h2 = relu(scatter(ew * (h1@W2)[src] -> dst) + b2)     K=128 (WMMA bf16x3)
//   L3: out = scatter(ew * (h2@W3)[src] -> dst) + b3          N=1   (scalar dot)
// Memory-bound: all intermediates live in d_ws (L2-resident); segment-sum via
// f32 global atomics, one wave per edge.
// ---------------------------------------------------------------------------

typedef __attribute__((ext_vector_type(16))) __bf16 v16bf;
typedef __attribute__((ext_vector_type(8)))  float  v8f;

static __device__ __forceinline__ unsigned short f32_to_bf16_rne(float f) {
    unsigned int u = __float_as_uint(f);
    unsigned int r = 0x7FFFu + ((u >> 16) & 1u);
    return (unsigned short)((u + r) >> 16);
}
static __device__ __forceinline__ float bf16u_to_f32(unsigned short h) {
    return __uint_as_float(((unsigned int)h) << 16);
}

// ---- L1 linear: h1lin[n,128] = x[n,2] @ W1[2,128] ---------------------------
__global__ void k_lin1(const float* __restrict__ x, const float* __restrict__ W1,
                       float* __restrict__ h1lin, int n_nodes) {
    int t = blockIdx.x * blockDim.x + threadIdx.x;
    int node = t >> 5;
    int f4   = (t & 31) << 2;
    if (node >= n_nodes) return;
    float x0 = x[(long)node * 2 + 0];
    float x1 = x[(long)node * 2 + 1];
    float4 w0 = *(const float4*)(W1 + f4);          // W1[0][f..f+3]
    float4 w1 = *(const float4*)(W1 + 128 + f4);    // W1[1][f..f+3]
    float4 o;
    o.x = x0 * w0.x + x1 * w1.x;
    o.y = x0 * w0.y + x1 * w1.y;
    o.z = x0 * w0.z + x1 * w1.z;
    o.w = x0 * w0.w + x1 * w1.w;
    *(float4*)(h1lin + (long)node * 128 + f4) = o;
}

// ---- init agg[n,feat] with broadcast bias (feat power of two, mult of 4) ----
__global__ void k_init_bias(float* __restrict__ dstv, const float* __restrict__ bias,
                            int n_nodes, int feat) {
    long idx   = (long)blockIdx.x * blockDim.x + threadIdx.x;  // one float4
    long total = (long)n_nodes * feat >> 2;
    if (idx >= total) return;
    int f4 = (int)((idx << 2) & (long)(feat - 1));
    float4 b = *(const float4*)(bias + f4);
    *(float4*)(dstv + (idx << 2)) = b;
}

// ---- scatter-add, 128 features: one wave per edge, 4 feats/lane -------------
__global__ void k_scatter128(const int* __restrict__ src, const int* __restrict__ dst,
                             const float* __restrict__ ew,
                             const float* __restrict__ h, float* __restrict__ agg,
                             int n_edges) {
    int t    = blockIdx.x * blockDim.x + threadIdx.x;
    int e    = t >> 5;
    int lane = t & 31;
    if (e >= n_edges) return;
    int s = src[e], d = dst[e];
    float w = ew[e];
    float4 v  = *(const float4*)(h + (long)s * 128 + (lane << 2));
    float* ap = agg + (long)d * 128 + (lane << 2);
    atomicAdd(ap + 0, w * v.x);
    atomicAdd(ap + 1, w * v.y);
    atomicAdd(ap + 2, w * v.z);
    atomicAdd(ap + 3, w * v.w);
}

// ---- scatter-add, 64 features: one wave per edge, 2 feats/lane --------------
__global__ void k_scatter64(const int* __restrict__ src, const int* __restrict__ dst,
                            const float* __restrict__ ew,
                            const float* __restrict__ h, float* __restrict__ agg,
                            int n_edges) {
    int t    = blockIdx.x * blockDim.x + threadIdx.x;
    int e    = t >> 5;
    int lane = t & 31;
    if (e >= n_edges) return;
    int s = src[e], d = dst[e];
    float w = ew[e];
    float2 v  = *(const float2*)(h + (long)s * 64 + (lane << 1));
    float* ap = agg + (long)d * 64 + (lane << 1);
    atomicAdd(ap + 0, w * v.x);
    atomicAdd(ap + 1, w * v.y);
}

// ---- relu + bf16 hi/lo split of agg1 (for bf16x3 WMMA GEMM) -----------------
__global__ void k_relu_split(const float* __restrict__ in,
                             unsigned short* __restrict__ hi,
                             unsigned short* __restrict__ lo,
                             long n_elems) {
    long idx = (long)blockIdx.x * blockDim.x + threadIdx.x;  // 2 elems/thread
    long i2  = idx << 1;
    if (i2 >= n_elems) return;
    float a = fmaxf(in[i2 + 0], 0.0f);
    float b = fmaxf(in[i2 + 1], 0.0f);
    unsigned short ah = f32_to_bf16_rne(a);
    unsigned short bh = f32_to_bf16_rne(b);
    unsigned short al = f32_to_bf16_rne(a - bf16u_to_f32(ah));
    unsigned short bl = f32_to_bf16_rne(b - bf16u_to_f32(bh));
    ((unsigned int*)hi)[idx] = (unsigned int)ah | ((unsigned int)bh << 16);
    ((unsigned int*)lo)[idx] = (unsigned int)al | ((unsigned int)bl << 16);
}

// ---- W2[128,64] -> transposed bf16 hi/lo: w2T[n*128 + k] --------------------
__global__ void k_w2_prep(const float* __restrict__ W2,
                          unsigned short* __restrict__ w2hT,
                          unsigned short* __restrict__ w2lT) {
    int t = blockIdx.x * blockDim.x + threadIdx.x;
    if (t >= 128 * 64) return;
    int k = t >> 6, n = t & 63;
    float v = W2[t];
    unsigned short h = f32_to_bf16_rne(v);
    unsigned short l = f32_to_bf16_rne(v - bf16u_to_f32(h));
    w2hT[n * 128 + k] = h;
    w2lT[n * 128 + k] = l;
}

// ---- L2 GEMM: h2lin[M=n_nodes,64] = relu(agg1)[M,128] @ W2[128,64] ----------
// One wave per 16x16 output tile; K=128 -> 4 steps of v_wmma_f32_16x16x32_bf16,
// 3 WMMAs per step (bf16x3 split: Ah*Bh + Ah*Bl + Al*Bh), f32 accumulate.
// A frag (16-bit 16x32): lane half h, elems 0-7 <-> K=k0+8h.., elems 8-15 <-> K=k0+16+8h..
// B frag (16-bit 32x16): col = n0+(lane&15), elems 0-7 <-> K=k0+8h.., elems 8-15 <-> K=k0+16+8h..
__global__ void k_gemm2_wmma(const unsigned short* __restrict__ a1h,
                             const unsigned short* __restrict__ a1l,
                             const unsigned short* __restrict__ w2hT,
                             const unsigned short* __restrict__ w2lT,
                             float* __restrict__ h2lin, int n_nodes) {
    int gt   = blockIdx.x * blockDim.x + threadIdx.x;
    int wave = gt >> 5;
    int lane = threadIdx.x & 31;
    int mtiles = (n_nodes + 15) >> 4;
    int mt = wave >> 2;          // 4 N-tiles (64 cols) per M-tile
    int nt = wave & 3;
    if (mt >= mtiles) return;    // wave-uniform exit: EXEC stays all-1s for WMMA

    const int m0   = mt << 4;
    const int n0   = nt << 4;
    const int half = lane >> 4;
    const int l15  = lane & 15;

    int arow = m0 + l15;
    if (arow >= n_nodes) arow = n_nodes - 1;   // clamp: padded rows never stored
    const unsigned short* ah_p = a1h + (long)arow * 128;
    const unsigned short* al_p = a1l + (long)arow * 128;
    const unsigned short* bh_p = w2hT + (long)(n0 + l15) * 128;
    const unsigned short* bl_p = w2lT + (long)(n0 + l15) * 128;

    union Frag { v16bf v; uint4 q[2]; };
    v8f acc = {};

#pragma unroll
    for (int ks = 0; ks < 4; ++ks) {
        const int k0 = ks * 32;
        Frag Ah, Al, Bh, Bl;
        Ah.q[0] = *(const uint4*)(ah_p + k0 + 8 * half);
        Ah.q[1] = *(const uint4*)(ah_p + k0 + 16 + 8 * half);
        Al.q[0] = *(const uint4*)(al_p + k0 + 8 * half);
        Al.q[1] = *(const uint4*)(al_p + k0 + 16 + 8 * half);
        Bh.q[0] = *(const uint4*)(bh_p + k0 + 8 * half);
        Bh.q[1] = *(const uint4*)(bh_p + k0 + 16 + 8 * half);
        Bl.q[0] = *(const uint4*)(bl_p + k0 + 8 * half);
        Bl.q[1] = *(const uint4*)(bl_p + k0 + 16 + 8 * half);
        // smallest terms first for accuracy
        acc = __builtin_amdgcn_wmma_f32_16x16x32_bf16(false, Al.v, false, Bh.v,
                                                      (short)0, acc, false, false);
        acc = __builtin_amdgcn_wmma_f32_16x16x32_bf16(false, Ah.v, false, Bl.v,
                                                      (short)0, acc, false, false);
        acc = __builtin_amdgcn_wmma_f32_16x16x32_bf16(false, Ah.v, false, Bh.v,
                                                      (short)0, acc, false, false);
    }

    // D layout: VGPR v -> row m0 + v + 8*half, col n0 + (lane&15)
    const int col = n0 + l15;
#pragma unroll
    for (int v = 0; v < 8; ++v) {
        int r = m0 + v + 8 * half;
        if (r < n_nodes) h2lin[(long)r * 64 + col] = acc[v];
    }
}

// ---- L3 linear: h3lin[n] = relu(agg2[n,:]) . W3[:,0] ------------------------
__global__ void k_lin3(const float* __restrict__ agg2, const float* __restrict__ W3,
                       float* __restrict__ h3lin, int n_nodes) {
    int i = blockIdx.x * blockDim.x + threadIdx.x;
    if (i >= n_nodes) return;
    const float* r = agg2 + (long)i * 64;
    float acc = 0.0f;
#pragma unroll
    for (int f = 0; f < 64; f += 4) {
        float4 v = *(const float4*)(r + f);
        float4 w = *(const float4*)(W3 + f);
        acc += fmaxf(v.x, 0.0f) * w.x + fmaxf(v.y, 0.0f) * w.y +
               fmaxf(v.z, 0.0f) * w.z + fmaxf(v.w, 0.0f) * w.w;
    }
    h3lin[i] = acc;
}

__global__ void k_init_out(float* __restrict__ out, const float* __restrict__ b3,
                           int n_nodes) {
    int i = blockIdx.x * blockDim.x + threadIdx.x;
    if (i >= n_nodes) return;
    out[i] = b3[0];
}

__global__ void k_scatter_out(const int* __restrict__ src, const int* __restrict__ dst,
                              const float* __restrict__ ew,
                              const float* __restrict__ h3, float* __restrict__ out,
                              int n_edges) {
    int e = blockIdx.x * blockDim.x + threadIdx.x;
    if (e >= n_edges) return;
    atomicAdd(out + dst[e], ew[e] * h3[src[e]]);
}

// ---------------------------------------------------------------------------
extern "C" void kernel_launch(void* const* d_in, const int* in_sizes, int n_in,
                              void* d_out, int out_size, void* d_ws, size_t ws_size,
                              hipStream_t stream) {
    (void)n_in; (void)out_size; (void)ws_size;
    const float* x  = (const float*)d_in[0];
    const int*   ei = (const int*)d_in[1];     // harness contract: integer -> int32
    const float* ew = (const float*)d_in[2];
    const float* W1 = (const float*)d_in[3];
    const float* b1 = (const float*)d_in[4];
    const float* W2 = (const float*)d_in[5];
    const float* b2 = (const float*)d_in[6];
    const float* W3 = (const float*)d_in[7];
    const float* b3 = (const float*)d_in[8];
    float* out = (float*)d_out;

    const int nn = in_sizes[0] / 2;   // x is [N, 2]
    const int ne = in_sizes[2];       // edge_weight count
    const int* src = ei;
    const int* dst = ei + ne;

    // --- workspace layout (aliased across pipeline stages) ---
    auto al = [](size_t v) { return (v + 255) & ~(size_t)255; };
    char* ws = (char*)d_ws;
    size_t off = 0;
    float* h1lin = (float*)(ws + off); off = al(off + (size_t)nn * 128 * 4);  // A
    float* agg1  = (float*)(ws + off); off = al(off + (size_t)nn * 128 * 4);  // B
    unsigned short* a1h = (unsigned short*)(ws + off); off = al(off + (size_t)nn * 128 * 2); // C
    unsigned short* a1l = (unsigned short*)(ws + off); off = al(off + (size_t)nn * 128 * 2); // D
    unsigned short* w2hT = (unsigned short*)(ws + off); off = al(off + 128 * 64 * 2);
    unsigned short* w2lT = (unsigned short*)(ws + off); off = al(off + 128 * 64 * 2);
    float* h2lin = h1lin;           // region A dead after scatter128
    float* agg2  = agg1;            // region B dead after relu_split
    float* h3lin = (float*)a1h;     // region C dead after gemm2

    const int B = 256;
    auto blocks = [](long t, int b) { return (int)((t + b - 1) / b); };

    // L1
    k_lin1<<<blocks((long)nn * 32, B), B, 0, stream>>>(x, W1, h1lin, nn);
    k_init_bias<<<blocks((long)nn * 32, B), B, 0, stream>>>(agg1, b1, nn, 128);
    k_scatter128<<<blocks((long)ne * 32, B), B, 0, stream>>>(src, dst, ew, h1lin, agg1, ne);
    k_relu_split<<<blocks((long)nn * 64, B), B, 0, stream>>>(agg1, a1h, a1l, (long)nn * 128);

    // L2 (WMMA GEMM)
    k_w2_prep<<<blocks(128 * 64, B), B, 0, stream>>>(W2, w2hT, w2lT);
    {
        long waves = (long)((nn + 15) >> 4) * 4;
        k_gemm2_wmma<<<blocks(waves * 32, B), B, 0, stream>>>(a1h, a1l, w2hT, w2lT, h2lin, nn);
    }
    k_init_bias<<<blocks((long)nn * 16, B), B, 0, stream>>>(agg2, b2, nn, 64);
    k_scatter64<<<blocks((long)ne * 32, B), B, 0, stream>>>(src, dst, ew, h2lin, agg2, ne);

    // L3
    k_lin3<<<blocks(nn, B), B, 0, stream>>>(agg2, W3, h3lin, nn);
    k_init_out<<<blocks(nn, B), B, 0, stream>>>(out, b3, nn);
    k_scatter_out<<<blocks(ne, B), B, 0, stream>>>(src, dst, ew, h3lin, out, ne);
}
